// RGG_46978352284517
// MI455X (gfx1250) — matrix operand order repaired and checked by reading, not compiled, not run on hardware
//
#include <hip/hip_runtime.h>
#include <hip/hip_bf16.h>

// ---------------------------------------------------------------------------
// ResGatedGraphConv GNN on MI455X (gfx1250, wave32, WMMA, async-to-LDS)
//   N=50000 nodes, E=800000 edges, HID=64, 4 layers, 512 graphs, OUT=12
// ---------------------------------------------------------------------------

#define N_NODES  50000
#define N_EDGES  800000
#define HID      64
#define OUTD     12
#define N_LAYERS 4
#define N_GRAPHS 512

typedef __attribute__((ext_vector_type(16))) _Float16     v16h;
typedef __attribute__((ext_vector_type(8)))  float        v8f;
typedef __attribute__((ext_vector_type(8)))  unsigned int v8u;

// Exact operand types for the gfx1250 async-to-LDS builtin (from hipcc
// diagnostics: param 0 is "int __attribute__((vector_size(16))) __device__*").
typedef int v4i_vs __attribute__((vector_size(16)));
typedef __attribute__((address_space(1))) v4i_vs* as1_v4i;
typedef __attribute__((address_space(3))) v4i_vs* as3_v4i;

#if defined(__has_builtin)
#if __has_builtin(__builtin_amdgcn_global_load_async_to_lds_b128)
#define HAVE_ASYNC_LDS 1
#endif
#endif

__device__ __forceinline__ void wait_async0() {
#if defined(__has_builtin)
#if __has_builtin(__builtin_amdgcn_s_wait_asynccnt)
    __builtin_amdgcn_s_wait_asynccnt(0);
    return;
#endif
#endif
    asm volatile("s_wait_asynccnt 0" ::: "memory");
}

// ---------------------------------------------------------------------------
// Pre-swizzle f32 weights into f16 WMMA B-fragment layout.
// Tile id = ((layer*4 + mat)*2 + kstep)*4 + ctile, 128 tiles, 32 lanes each.
// B (32x16 f16) lane layout: col n = lane&15, rows K = (lane>>4)*16 + j,
// packed 2 halves per dword, 8 dwords per lane (32B, coalesced).
// ---------------------------------------------------------------------------
__global__ void RGG_prepack_w(const float* __restrict__ Wk,
                              const float* __restrict__ Wq,
                              const float* __restrict__ Wv,
                              const float* __restrict__ Ws,
                              unsigned int* __restrict__ packed) {
    int tile = blockIdx.x;          // 0..127
    int lane = threadIdx.x;         // 0..31
    int c   =  tile        & 3;
    int s   = (tile >> 2)  & 1;
    int mat = (tile >> 3)  & 3;
    int l   =  tile >> 5;
    const float* W = (mat == 0 ? Wk : mat == 1 ? Wq : mat == 2 ? Wv : Ws)
                     + (size_t)l * HID * HID;
    int n  = (c << 4) + (lane & 15);
    int rb = ((lane >> 4) << 4) + (s << 5);   // K row base for this lane
    unsigned int* o = packed + (size_t)tile * 256 + lane * 8;
#pragma unroll
    for (int d = 0; d < 8; ++d) {
        int k0 = rb + 2 * d;
        _Float16 h0 = (_Float16)W[(size_t)k0 * HID + n];
        _Float16 h1 = (_Float16)W[(size_t)(k0 + 1) * HID + n];
        unsigned int u0 = (unsigned int)__builtin_bit_cast(unsigned short, h0);
        unsigned int u1 = (unsigned int)__builtin_bit_cast(unsigned short, h1);
        o[d] = (u1 << 16) | u0;
    }
}

// ---------------------------------------------------------------------------
// Node transform: K = X Wk + bk, Q = X Wq + bq, V = X Wv + bv, agg = X Ws + bs
// One block = 4 waves; wave w handles matrix w, full 16x64 tile, 8 WMMAs.
// Layer weights (32KB) + X tile (4KB) staged in LDS (async-to-LDS if avail).
// ---------------------------------------------------------------------------
__global__ void __launch_bounds__(128)
RGG_gemm_kqvs(const float* __restrict__ X,
              const unsigned int* __restrict__ packedL,  // this layer: 8192 dwords
              const float* __restrict__ bk, const float* __restrict__ bq,
              const float* __restrict__ bv, const float* __restrict__ bs,
              float* __restrict__ kout, float* __restrict__ qout,
              float* __restrict__ vout, float* __restrict__ aggout) {
    __shared__ __attribute__((aligned(16))) unsigned int sW[8192];  // 32 KB
    __shared__ __attribute__((aligned(16))) float        sX[16 * HID]; // 4 KB

    const int tid  = threadIdx.x;
    const int wave = tid >> 5;             // 0..3 == matrix id
    const int lane = tid & 31;
    const int m0   = blockIdx.x << 4;      // 16-row tile (3125 tiles, exact)

    // ---- Stage layer weights + X tile into LDS -----------------------------
#if HAVE_ASYNC_LDS
    {
#pragma unroll
        for (int i = 0; i < 16; ++i) {       // 8192 dw = 2048 b128 / 128 thr
            int idx = (tid + i * 128) * 4;
            __builtin_amdgcn_global_load_async_to_lds_b128(
                (as1_v4i)(packedL + idx), (as3_v4i)(sW + idx), 0, 0);
        }
#pragma unroll
        for (int i = 0; i < 2; ++i) {        // 1024 dw = 256 b128 / 128 thr
            int idx = (tid + i * 128) * 4;
            __builtin_amdgcn_global_load_async_to_lds_b128(
                (as1_v4i)(X + (size_t)m0 * HID + idx), (as3_v4i)(sX + idx), 0, 0);
        }
        wait_async0();
    }
#else
#pragma unroll
    for (int i = 0; i < 16; ++i) {
        int idx = (tid + i * 128) * 4;
        *(uint4*)(sW + idx) = *(const uint4*)(packedL + idx);
    }
#pragma unroll
    for (int i = 0; i < 2; ++i) {
        int idx = (tid + i * 128) * 4;
        *(float4*)(sX + idx) = *(const float4*)(X + (size_t)m0 * HID + idx);
    }
#endif
    __syncthreads();

    const float* bias = (wave == 0) ? bk : (wave == 1) ? bq : (wave == 2) ? bv : bs;
    float*       out  = (wave == 0) ? kout : (wave == 1) ? qout : (wave == 2) ? vout : aggout;

    // Accumulators seeded with bias (bias depends only on N -> broadcast over M)
    v8f acc[4];
#pragma unroll
    for (int c = 0; c < 4; ++c) {
        float bn = bias[(c << 4) + (lane & 15)];
#pragma unroll
        for (int r = 0; r < 8; ++r) acc[c][r] = bn;
    }

    const float* xrow = sX + (lane & 15) * HID;

#pragma unroll
    for (int s = 0; s < 2; ++s) {
        // A fragment (16x32 f16): lane 0-15 rows M=lane,   K = s*32 + {0..15}
        //                         lane16-31 rows M=lane-16, K = s*32 + {16..31}
        const int kb = ((lane >> 4) << 4) + (s << 5);
        float4 x0 = *(const float4*)(xrow + kb);
        float4 x1 = *(const float4*)(xrow + kb + 4);
        float4 x2 = *(const float4*)(xrow + kb + 8);
        float4 x3 = *(const float4*)(xrow + kb + 12);
        v16h a;
        a[0]  = (_Float16)x0.x; a[1]  = (_Float16)x0.y;
        a[2]  = (_Float16)x0.z; a[3]  = (_Float16)x0.w;
        a[4]  = (_Float16)x1.x; a[5]  = (_Float16)x1.y;
        a[6]  = (_Float16)x1.z; a[7]  = (_Float16)x1.w;
        a[8]  = (_Float16)x2.x; a[9]  = (_Float16)x2.y;
        a[10] = (_Float16)x2.z; a[11] = (_Float16)x2.w;
        a[12] = (_Float16)x3.x; a[13] = (_Float16)x3.y;
        a[14] = (_Float16)x3.z; a[15] = (_Float16)x3.w;

#pragma unroll
        for (int c = 0; c < 4; ++c) {
            const v8u braw = *(const v8u*)(sW +
                (size_t)(((wave << 1) + s) * 4 + c) * 256 + lane * 8);
            v16h bfrag = __builtin_bit_cast(v16h, braw);
            acc[c] = __builtin_amdgcn_wmma_f32_16x16x32_f16(
                false, a, false, bfrag, (short)0, acc[c], false, false);
        }
    }

    // Store D: VGPR r -> row m0 + (lane>=16 ? 8 : 0) + r, col = c*16 + (lane&15)
    const int rhi = (lane >> 4) << 3;
#pragma unroll
    for (int c = 0; c < 4; ++c) {
        const int col = (c << 4) + (lane & 15);
#pragma unroll
        for (int r = 0; r < 8; ++r)
            out[(size_t)(m0 + rhi + r) * HID + col] = acc[c][r];
    }
}

// ---------------------------------------------------------------------------
// Edge phase: one wave per edge, 2 features per lane.
//   gate = sigmoid(k[dst] + q[src]);  agg[dst] += gate * v[src]
// ---------------------------------------------------------------------------
__global__ void __launch_bounds__(256)
RGG_edge(const float* __restrict__ k, const float* __restrict__ q,
         const float* __restrict__ v, const int* __restrict__ ei,
         float* __restrict__ agg) {
    int e = blockIdx.x * 8 + (threadIdx.x >> 5);
    if (e >= N_EDGES) return;
    int lane = threadIdx.x & 31;
    int src = ei[e];
    int dst = ei[N_EDGES + e];
    int f = lane << 1;
    float2 kd = *(const float2*)(k + (size_t)dst * HID + f);
    float2 qs = *(const float2*)(q + (size_t)src * HID + f);
    float2 vs = *(const float2*)(v + (size_t)src * HID + f);
    float g0 = 1.0f / (1.0f + __expf(-(kd.x + qs.x)));
    float g1 = 1.0f / (1.0f + __expf(-(kd.y + qs.y)));
    float* ap = agg + (size_t)dst * HID + f;
    atomicAdd(ap,     g0 * vs.x);
    atomicAdd(ap + 1, g1 * vs.y);
}

// ---------------------------------------------------------------------------
// Exact GELU: x -> 0.5 x (1 + erf(x/sqrt(2)));  agg -> xnext
// ---------------------------------------------------------------------------
__global__ void RGG_gelu(const float* __restrict__ agg, float* __restrict__ xout) {
    int i = blockIdx.x * blockDim.x + threadIdx.x;   // N_NODES*HID threads exact
    float t = agg[i];
    xout[i] = 0.5f * t * (1.0f + erff(t * 0.70710678118654752f));
}

__global__ void RGG_zero(float* __restrict__ p, int n) {
    int i = blockIdx.x * blockDim.x + threadIdx.x;
    if (i < n) p[i] = 0.0f;
}

// ---------------------------------------------------------------------------
// Global mean pool (segment sums via atomics): one wave per node
// ---------------------------------------------------------------------------
__global__ void __launch_bounds__(256)
RGG_pool(const float* __restrict__ x, const int* __restrict__ batch,
         float* __restrict__ sums, float* __restrict__ cnts) {
    int node = blockIdx.x * 8 + (threadIdx.x >> 5);
    if (node >= N_NODES) return;
    int lane = threadIdx.x & 31;
    int b = batch[node];
    int f = lane << 1;
    float2 xv = *(const float2*)(x + (size_t)node * HID + f);
    atomicAdd(sums + (size_t)b * HID + f,     xv.x);
    atomicAdd(sums + (size_t)b * HID + f + 1, xv.y);
    if (lane == 0) atomicAdd(cnts + b, 1.0f);
}

// ---------------------------------------------------------------------------
// MLP head: pooled -> Linear(64) -> ReLU -> Linear(12); one block per graph
// ---------------------------------------------------------------------------
__global__ void __launch_bounds__(64)
RGG_head(const float* __restrict__ sums, const float* __restrict__ cnts,
         const float* __restrict__ w1, const float* __restrict__ b1,
         const float* __restrict__ w2, const float* __restrict__ b2,
         float* __restrict__ out) {
    __shared__ float pooled[HID];
    __shared__ float h[HID];
    int b = blockIdx.x, t = threadIdx.x;
    float inv = 1.0f / fmaxf(cnts[b], 1.0f);
    pooled[t] = sums[(size_t)b * HID + t] * inv;
    __syncthreads();
    float acc = b1[t];
#pragma unroll 8
    for (int kk = 0; kk < HID; ++kk) acc += pooled[kk] * w1[(size_t)kk * HID + t];
    h[t] = fmaxf(acc, 0.0f);
    __syncthreads();
    if (t < OUTD) {
        float o = b2[t];
#pragma unroll 8
        for (int kk = 0; kk < HID; ++kk) o += h[kk] * w2[(size_t)kk * OUTD + t];
        out[(size_t)b * OUTD + t] = o;
    }
}

// ---------------------------------------------------------------------------
// Host-side orchestration
// ---------------------------------------------------------------------------
extern "C" void kernel_launch(void* const* d_in, const int* in_sizes, int n_in,
                              void* d_out, int out_size, void* d_ws, size_t ws_size,
                              hipStream_t stream) {
    const float* x_in   = (const float*)d_in[0];
    const int*   ei     = (const int*)  d_in[1];
    const int*   batch  = (const int*)  d_in[2];
    const float* Wk     = (const float*)d_in[4];
    const float* bk     = (const float*)d_in[5];
    const float* Wq     = (const float*)d_in[6];
    const float* bq     = (const float*)d_in[7];
    const float* Wv     = (const float*)d_in[8];
    const float* bv     = (const float*)d_in[9];
    const float* Ws     = (const float*)d_in[10];
    const float* bs     = (const float*)d_in[11];
    const float* w1     = (const float*)d_in[12];
    const float* b1     = (const float*)d_in[13];
    const float* w2     = (const float*)d_in[14];
    const float* b2     = (const float*)d_in[15];
    float* out = (float*)d_out;

    // Workspace layout (bytes)
    const size_t NB = (size_t)N_NODES * HID * sizeof(float);   // 12.8 MB
    char* ws = (char*)d_ws;
    unsigned int* packed = (unsigned int*)ws;                  // 128 KB
    float* xbuf = (float*)(ws + 131072);
    float* kbuf = (float*)(ws + 131072 + 1 * NB);
    float* qbuf = (float*)(ws + 131072 + 2 * NB);
    float* vbuf = (float*)(ws + 131072 + 3 * NB);
    float* agg  = (float*)(ws + 131072 + 4 * NB);
    float* sums = (float*)(ws + 131072 + 5 * NB);              // 512*64 f32
    float* cnts = (float*)(ws + 131072 + 5 * NB + (size_t)N_GRAPHS * HID * sizeof(float));

    // 1) Pre-swizzle all layer weights into WMMA B-fragment layout (f16)
    RGG_prepack_w<<<N_LAYERS * 4 * 2 * 4, 32, 0, stream>>>(Wk, Wq, Wv, Ws, packed);

    // 2) Layers
    const float* xptr = x_in;
    for (int l = 0; l < N_LAYERS; ++l) {
        const unsigned int* pl = packed + (size_t)l * 4 * 8 * 256;  // 4 mats * 8 tiles * 256 dw
        RGG_gemm_kqvs<<<N_NODES / 16, 128, 0, stream>>>(
            xptr, pl,
            bk + (size_t)l * HID, bq + (size_t)l * HID,
            bv + (size_t)l * HID, bs + (size_t)l * HID,
            kbuf, qbuf, vbuf, agg);
        RGG_edge<<<N_EDGES / 8, 256, 0, stream>>>(kbuf, qbuf, vbuf, ei, agg);
        RGG_gelu<<<(N_NODES * HID) / 256, 256, 0, stream>>>(agg, xbuf);
        xptr = xbuf;
    }

    // 3) Pool + head
    RGG_zero<<<(N_GRAPHS * HID + N_GRAPHS + 255) / 256, 256, 0, stream>>>(
        sums, N_GRAPHS * HID + N_GRAPHS);   // sums and cnts are contiguous
    RGG_pool<<<N_NODES / 8 + 1, 256, 0, stream>>>(xbuf, batch, sums, cnts);
    RGG_head<<<N_GRAPHS, 64, 0, stream>>>(sums, cnts, w1, b1, w2, b2, out);
}
